// self_attention_30709016166855
// MI455X (gfx1250) — matrix operand order repaired and checked by reading, not compiled, not run on hardware
//
#include <hip/hip_runtime.h>

// ---------------------------------------------------------------------------
// MI455X (gfx1250, wave32) self-attention block.
// GEMMs: v_wmma_f32_16x16x32_bf16 (fp32 accum, bf16 operands).
// TDM (tensor_load_to_lds) stages shared tiles into LDS (K2b, K3).
// sched_barrier(0) separates fragment-load clauses from WMMA bursts so the
// backend pipelines loads against matrix issue instead of wait-serializing.
// ---------------------------------------------------------------------------

typedef __attribute__((ext_vector_type(16))) __bf16 v16bf;
typedef __attribute__((ext_vector_type(8)))  float  v8f;
typedef __attribute__((ext_vector_type(4)))  unsigned int u32x4;
typedef __attribute__((ext_vector_type(8)))  int i32x8;
typedef __attribute__((ext_vector_type(4)))  int i32x4;

#define SCHED_FENCE() __builtin_amdgcn_sched_barrier(0)

union Frag16 {
  v16bf v;
  uint4 q[2];
  __bf16 b[16];
};

__device__ __forceinline__ unsigned short f32_to_bf16(float f) {
  union { __bf16 b; unsigned short u; } c;
  c.b = (__bf16)f;            // native v_cvt on gfx1250, RNE
  return c.u;
}

__device__ __forceinline__ v8f zero8() {
  v8f z;
#pragma unroll
  for (int i = 0; i < 8; ++i) z[i] = 0.0f;
  return z;
}

// A fragment (16x32, M x K), contiguous-K order (same as load_b_kt), not ISA table order:
// lane&15 = row; half=lane>>4: 16 contiguous K at [16*hf .. +15]
__device__ __forceinline__ v16bf load_a_bf16(const unsigned short* src,
                                             int ld, int row0, int k0, int lane) {
  const int r = lane & 15, hf = lane >> 4;
  const unsigned short* p = src + (size_t)(row0 + r) * ld + k0;
  Frag16 f;
  f.q[0] = *(const uint4*)(p + 16 * hf);
  f.q[1] = *(const uint4*)(p + 16 * hf + 8);
  return f.v;
}

// B fragment (32x16, K x N) from K-major matrix BT[n][k]:
// lane&15 = col; 16 contiguous bf16 at k0 + 16*half
__device__ __forceinline__ v16bf load_b_kt(const unsigned short* bt,
                                           int ld, int n0, int k0, int lane) {
  const int r = lane & 15, hf = lane >> 4;
  const unsigned short* p = bt + (size_t)(n0 + r) * ld + k0 + 16 * hf;
  Frag16 f;
  f.q[0] = *(const uint4*)(p);
  f.q[1] = *(const uint4*)(p + 8);
  return f.v;
}

__device__ __forceinline__ v8f wmma_bf16(v16bf a, v16bf b, v8f c) {
  return __builtin_amdgcn_wmma_f32_16x16x32_bf16(false, a, false, b, (short)0, c,
                                                 false, false);
}

// ---------------------------------------------------------------------------
// Tensor Data Mover: 2-D tile load (bf16 elements) via D# descriptor
// (ISA 08_async_tensor §8.3/§8.4; groups 2/3 zero => dims 3..5 unused).
// ---------------------------------------------------------------------------
__device__ __forceinline__ unsigned lds_off(const void* p) {
  // generic LDS pointer: addr[31:0] is the wave-relative LDS byte offset
  return (unsigned)(unsigned long long)p;
}

__device__ __forceinline__ void tdm_load_2d(unsigned ldsByteOff, const void* gptr,
                                            unsigned rowElems,   // tensor_dim0 == stride
                                            unsigned tensRows,   // tensor_dim1
                                            unsigned tileD0, unsigned tileD1) {
  unsigned long long ga = (unsigned long long)gptr;
  u32x4 g0;
  g0[0] = 1u;                                             // count=1, user mode
  g0[1] = ldsByteOff;                                     // lds_addr
  g0[2] = (unsigned)(ga & 0xFFFFFFFFu);                   // global_addr[31:0]
  g0[3] = (unsigned)((ga >> 32) & 0x1FFFFFFu) | (2u << 30); // [56:32] | type=2

  unsigned long long s0 = rowElems;                       // dim0 stride (elems)
  i32x8 g1;
  g1[0] = (int)(1u << 16);                                // data_size=1 (2 bytes)
  g1[1] = (int)((rowElems & 0xFFFFu) << 16);              // tensor_dim0[15:0]
  g1[2] = (int)(((rowElems >> 16) & 0xFFFFu) |            // tensor_dim0[31:16]
                ((tensRows & 0xFFFFu) << 16));            // tensor_dim1[15:0]
  g1[3] = (int)(((tensRows >> 16) & 0xFFFFu) |            // tensor_dim1[31:16]
                ((tileD0 & 0xFFFFu) << 16));              // tile_dim0
  g1[4] = (int)(tileD1 & 0xFFFFu);                        // tile_dim1 (tile_dim2=0)
  g1[5] = (int)(unsigned)(s0 & 0xFFFFFFFFull);            // dim0_stride[31:0]
  g1[6] = (int)(unsigned)((s0 >> 32) & 0xFFFFull);        // dim0_stride[47:32]
  g1[7] = 0;                                              // dim1_stride (unused 2D)

  i32x4 z4 = {0, 0, 0, 0};
#if defined(__clang_major__) && (__clang_major__ >= 23)
  i32x8 z8 = {0, 0, 0, 0, 0, 0, 0, 0};
  __builtin_amdgcn_tensor_load_to_lds(g0, g1, z4, z4, z8, 0);
#else
  __builtin_amdgcn_tensor_load_to_lds(g0, g1, z4, z4, 0);
#endif
}

// ---------------------------------------------------------------------------
// K0a: weight convert + transpose  W[K][Nc] f32 -> WT[Nc][K] bf16
// ---------------------------------------------------------------------------
__global__ void wtrans_kernel(const float* __restrict__ src,
                              unsigned short* __restrict__ dst, int K, int Nc) {
  int idx = blockIdx.x * blockDim.x + threadIdx.x;
  if (idx >= K * Nc) return;
  int n = idx / K;
  int k = idx - n * K;
  dst[idx] = f32_to_bf16(src[(size_t)k * Nc + n]);
}

// ---------------------------------------------------------------------------
// K0b: x f32 -> bf16 (streaming; 4 elements/thread)
// ---------------------------------------------------------------------------
__global__ __launch_bounds__(256) void xconv_kernel(const float* __restrict__ x,
                                                    unsigned short* __restrict__ Xb) {
  int idx = blockIdx.x * blockDim.x + threadIdx.x;   // quad index, 8388608 total
  float4 v = ((const float4*)x)[idx];
  uint2 o;
  o.x = (unsigned)f32_to_bf16(v.x) | ((unsigned)f32_to_bf16(v.y) << 16);
  o.y = (unsigned)f32_to_bf16(v.z) | ((unsigned)f32_to_bf16(v.w) << 16);
  ((uint2*)Xb)[idx] = o;
}

// ---------------------------------------------------------------------------
// K1: fused projections. Xb[32768,1024]bf16 -> F[32768,128], G[32768,128],
//     Ht[b][w*512+c][h] (H pre-transposed per batch for the K=64 GEMM).
// grid (512, 6), block 128 (4 waves). Each wave: 16 rows x 128 cols.
// ---------------------------------------------------------------------------
__global__ __launch_bounds__(128) void proj_kernel(
    const unsigned short* __restrict__ Xb,
    const unsigned short* __restrict__ WfT, const float* __restrict__ bfv,
    const unsigned short* __restrict__ WgT, const float* __restrict__ bgv,
    const unsigned short* __restrict__ WhT, const float* __restrict__ bhv,
    unsigned short* __restrict__ F, unsigned short* __restrict__ G,
    unsigned short* __restrict__ Ht) {
  const int tid = threadIdx.x;
  const int lane = tid & 31, wave = tid >> 5;
  const int r = lane & 15, hf = lane >> 4;
  const int row0 = blockIdx.x * 64 + wave * 16;
  const int cb = blockIdx.y;

  const unsigned short* WT;
  const float* bias;
  int colbase = 0;
  if (cb == 0)      { WT = WfT; bias = bfv; }
  else if (cb == 1) { WT = WgT; bias = bgv; }
  else {
    WT = WhT + (size_t)(cb - 2) * 128 * 1024;
    bias = bhv + (cb - 2) * 128;
    colbase = (cb - 2) * 128;
  }

  v8f acc[8];
#pragma unroll
  for (int i = 0; i < 8; ++i) acc[i] = zero8();

  for (int kk = 0; kk < 1024; kk += 32) {
    if (kk + 32 < 1024)
      __builtin_prefetch(Xb + (size_t)(row0 + r) * 1024 + kk + 32, 0, 1);
    // phase 1: issue all fragment loads (clauses, distinct regs)
    v16bf a = load_a_bf16(Xb, 1024, row0, kk, lane);
    v16bf bfr[8];
#pragma unroll
    for (int nb = 0; nb < 8; ++nb) bfr[nb] = load_b_kt(WT, 1024, nb * 16, kk, lane);
    SCHED_FENCE();
    // phase 2: WMMA burst; waits degrade as in-order loads retire
#pragma unroll
    for (int nb = 0; nb < 8; ++nb) acc[nb] = wmma_bf16(a, bfr[nb], acc[nb]);
  }

  if (cb < 2) {
    unsigned short* out = (cb == 0) ? F : G;
#pragma unroll
    for (int nb = 0; nb < 8; ++nb) {
      float bv = bias[nb * 16 + r];
#pragma unroll
      for (int v = 0; v < 8; ++v) {
        int rowG = row0 + v + 8 * hf;
        out[(size_t)rowG * 128 + nb * 16 + r] = f32_to_bf16(acc[nb][v] + bv);
      }
    }
  } else {
#pragma unroll
    for (int nb = 0; nb < 8; ++nb) {
      float bv = bias[nb * 16 + r];
      int c = colbase + nb * 16 + r;
#pragma unroll
      for (int v = 0; v < 8; ++v) {
        int rowG = row0 + v + 8 * hf;           // pixel = (b,h,w)
        int b_ = rowG >> 12;
        int h_ = (rowG >> 6) & 63;
        int w_ = rowG & 63;
        Ht[(size_t)b_ * 2097152 + (size_t)(w_ * 512 + c) * 64 + h_] =
            f32_to_bf16(acc[nb][v] + bv);
      }
    }
  }
}

// ---------------------------------------------------------------------------
// K2a: per-batch scores S = (G Fᵀ)/sqrt(128), softmax rows, BetaT[b][j][i].
// grid 8, block 128 (4 waves). Wave w: rows i = 16w..16w+15, all 64 j, K=8192.
// ---------------------------------------------------------------------------
__global__ __launch_bounds__(128) void attn_scores_kernel(
    const unsigned short* __restrict__ F, const unsigned short* __restrict__ G,
    unsigned short* __restrict__ BetaT) {
  __shared__ float S[64][64];
  const int b = blockIdx.x;
  const int tid = threadIdx.x, lane = tid & 31, wave = tid >> 5;
  const int r = lane & 15, hf = lane >> 4;
  const unsigned short* Fb = F + (size_t)b * 524288;   // [64][8192]
  const unsigned short* Gb = G + (size_t)b * 524288;   // [64][8192]

  v8f acc[4];
#pragma unroll
  for (int i = 0; i < 4; ++i) acc[i] = zero8();

  for (int kk = 0; kk < 8192; kk += 32) {
    v16bf a = load_a_bf16(Gb, 8192, wave * 16, kk, lane);
    v16bf bfr[4];
#pragma unroll
    for (int jb = 0; jb < 4; ++jb) bfr[jb] = load_b_kt(Fb, 8192, jb * 16, kk, lane);
    SCHED_FENCE();
#pragma unroll
    for (int jb = 0; jb < 4; ++jb) acc[jb] = wmma_bf16(a, bfr[jb], acc[jb]);
  }

  const float scale = 0.0883883476483184f;   // 1/sqrt(128)
#pragma unroll
  for (int jb = 0; jb < 4; ++jb)
#pragma unroll
    for (int v = 0; v < 8; ++v)
      S[wave * 16 + v + 8 * hf][jb * 16 + r] = acc[jb][v] * scale;
  __syncthreads();

  if (tid < 64) {
    const int i = tid;
    float mx = -3.4e38f;
    for (int j = 0; j < 64; ++j) mx = fmaxf(mx, S[i][j]);
    float sum = 0.0f;
    for (int j = 0; j < 64; ++j) sum += __expf(S[i][j] - mx);
    float inv = 1.0f / sum;
    for (int j = 0; j < 64; ++j)
      BetaT[(size_t)b * 4096 + j * 64 + i] = f32_to_bf16(__expf(S[i][j] - mx) * inv);
  }
}

// ---------------------------------------------------------------------------
// K2b: O = betaᵀ · H (per batch [64,64]x[64,32768], K=64).
// Shared tiles (128x64 Ht chunk + 64x64 betaT) staged via TDM into LDS.
// grid (256, 8), block 128 (4 waves).
// ---------------------------------------------------------------------------
__global__ __launch_bounds__(128) void attn_apply_kernel(
    const unsigned short* __restrict__ BetaT, const unsigned short* __restrict__ Ht,
    unsigned short* __restrict__ O) {
  __shared__ alignas(16) unsigned short sHt[128 * 64];    // 16 KB
  __shared__ alignas(16) unsigned short sBeta[64 * 64];   // 8 KB
  const int b = blockIdx.y;
  const int n0 = blockIdx.x * 128;
  const int tid = threadIdx.x, lane = tid & 31, wave = tid >> 5;
  const int r = lane & 15, hf = lane >> 4;

  if (wave == 0) {
    tdm_load_2d(lds_off(sHt), Ht + (size_t)b * 2097152 + (size_t)n0 * 64,
                /*rowElems=*/64, /*tensRows=*/128, /*tile=*/64, 128);
    tdm_load_2d(lds_off(sBeta), BetaT + (size_t)b * 4096,
                /*rowElems=*/64, /*tensRows=*/64, /*tile=*/64, 64);
    __builtin_amdgcn_s_wait_tensorcnt((short)0);
  }
  __syncthreads();

  v8f acc[8];
#pragma unroll
  for (int i = 0; i < 8; ++i) acc[i] = zero8();

#pragma unroll
  for (int kk = 0; kk < 64; kk += 32) {
    v16bf a = load_a_bf16(sBeta, 64, wave * 16, kk, lane);
    v16bf bfr[8];
#pragma unroll
    for (int nb = 0; nb < 8; ++nb) bfr[nb] = load_b_kt(sHt, 64, nb * 16, kk, lane);
    SCHED_FENCE();
#pragma unroll
    for (int nb = 0; nb < 8; ++nb) acc[nb] = wmma_bf16(a, bfr[nb], acc[nb]);
  }

#pragma unroll
  for (int nb = 0; nb < 8; ++nb)
#pragma unroll
    for (int v = 0; v < 8; ++v) {
      int j = wave * 16 + v + 8 * hf;
      int n = n0 + nb * 16 + r;
      O[(size_t)b * 2097152 + (size_t)j * 32768 + n] = f32_to_bf16(acc[nb][v]);
    }
}

// ---------------------------------------------------------------------------
// K3: Y = LN(0.002*relu(O·Wl + bl) + x). grid 2048 (16 rows), block 256
// (8 waves; wave w -> cols 128w..128w+127). The 16x512 O A-tile (contiguous
// 16 KB) is staged once via TDM; all 8 waves share it from LDS.
// LN stats via ds_add_f32 atomics.
// ---------------------------------------------------------------------------
__global__ __launch_bounds__(256) void out_kernel(
    const unsigned short* __restrict__ O, const unsigned short* __restrict__ WlT,
    const float* __restrict__ bl, const float* __restrict__ x,
    const float* __restrict__ gamma, const float* __restrict__ betaLN,
    float* __restrict__ out) {
  __shared__ alignas(16) unsigned short sO[16 * 512];   // 16 KB
  __shared__ float ssum[16];
  __shared__ float ssq[16];
  const int tid = threadIdx.x, lane = tid & 31, wave = tid >> 5;
  const int r = lane & 15, hf = lane >> 4;
  const int row0 = blockIdx.x * 16;

  if (tid < 16) { ssum[tid] = 0.0f; ssq[tid] = 0.0f; }
  if (wave == 0) {
    tdm_load_2d(lds_off(sO), O + (size_t)row0 * 512,
                /*rowElems=*/512, /*tensRows=*/16, /*tile=*/512, 16);
    __builtin_amdgcn_s_wait_tensorcnt((short)0);
  }
  __syncthreads();

  v8f acc[8];
#pragma unroll
  for (int i = 0; i < 8; ++i) acc[i] = zero8();

  for (int kk = 0; kk < 512; kk += 32) {
    v16bf a = load_a_bf16(sO, 512, 0, kk, lane);
    v16bf bfr[8];
#pragma unroll
    for (int nb = 0; nb < 8; ++nb)
      bfr[nb] = load_b_kt(WlT, 512, wave * 128 + nb * 16, kk, lane);
    SCHED_FENCE();
#pragma unroll
    for (int nb = 0; nb < 8; ++nb) acc[nb] = wmma_bf16(a, bfr[nb], acc[nb]);
  }

  // epilogue: y = 0.002*relu(gemm + bias) + x ; accumulate LN partials
  float s1[8], s2[8];
#pragma unroll
  for (int v = 0; v < 8; ++v) { s1[v] = 0.0f; s2[v] = 0.0f; }
#pragma unroll
  for (int nb = 0; nb < 8; ++nb) {
    int n = wave * 128 + nb * 16 + r;
    float bv = bl[n];
#pragma unroll
    for (int v = 0; v < 8; ++v) {
      int rowG = row0 + v + 8 * hf;
      float y = 0.002f * fmaxf(acc[nb][v] + bv, 0.0f) + x[(size_t)rowG * 1024 + n];
      acc[nb][v] = y;
      s1[v] += y;
      s2[v] += y * y;
    }
  }
#pragma unroll
  for (int v = 0; v < 8; ++v) {
    int m = v + 8 * hf;
    atomicAdd(&ssum[m], s1[v]);
    atomicAdd(&ssq[m], s2[v]);
  }
  __syncthreads();

#pragma unroll
  for (int nb = 0; nb < 8; ++nb) {
    int n = wave * 128 + nb * 16 + r;
    float gm = gamma[n], bt = betaLN[n];
#pragma unroll
    for (int v = 0; v < 8; ++v) {
      int m = v + 8 * hf;
      float mean = ssum[m] * (1.0f / 1024.0f);
      float var  = ssq[m] * (1.0f / 1024.0f) - mean * mean;
      float rstd = rsqrtf(var + 1e-6f);
      out[(size_t)(row0 + m) * 1024 + n] = gm * (acc[nb][v] - mean) * rstd + bt;
    }
  }
}

// ---------------------------------------------------------------------------
// Launch
// ---------------------------------------------------------------------------
extern "C" void kernel_launch(void* const* d_in, const int* in_sizes, int n_in,
                              void* d_out, int out_size, void* d_ws, size_t ws_size,
                              hipStream_t stream) {
  const float* x     = (const float*)d_in[0];
  const float* Wf    = (const float*)d_in[1];
  const float* bfv   = (const float*)d_in[2];
  const float* Wg    = (const float*)d_in[3];
  const float* bgv   = (const float*)d_in[4];
  const float* Wh    = (const float*)d_in[5];
  const float* bhv   = (const float*)d_in[6];
  const float* Wl    = (const float*)d_in[7];
  const float* blv   = (const float*)d_in[8];
  const float* gamma = (const float*)d_in[9];
  const float* betaL = (const float*)d_in[10];

  // workspace layout (bf16 elements)
  unsigned short* p = (unsigned short*)d_ws;
  unsigned short* WfT   = p; p += 128 * 1024;
  unsigned short* WgT   = p; p += 128 * 1024;
  unsigned short* WhT   = p; p += 512 * 1024;
  unsigned short* WlT   = p; p += 1024 * 512;
  unsigned short* F     = p; p += (size_t)32768 * 128;
  unsigned short* G     = p; p += (size_t)32768 * 128;
  unsigned short* Ht    = p; p += (size_t)8 * 2097152;
  unsigned short* BetaT = p; p += (size_t)8 * 4096;
  unsigned short* Xb    = p;            // [32768,1024] bf16, dead after proj
  unsigned short* O     = p;            // [32768,512]  bf16, aliases Xb region
  p += (size_t)32768 * 1024;

  wtrans_kernel<<<(128 * 1024 + 255) / 256, 256, 0, stream>>>(Wf, WfT, 1024, 128);
  wtrans_kernel<<<(128 * 1024 + 255) / 256, 256, 0, stream>>>(Wg, WgT, 1024, 128);
  wtrans_kernel<<<(512 * 1024 + 255) / 256, 256, 0, stream>>>(Wh, WhT, 1024, 512);
  wtrans_kernel<<<(512 * 1024 + 255) / 256, 256, 0, stream>>>(Wl, WlT, 512, 1024);
  xconv_kernel<<<32768, 256, 0, stream>>>(x, Xb);

  proj_kernel<<<dim3(512, 6), 128, 0, stream>>>(Xb, WfT, bfv, WgT, bgv, WhT, bhv,
                                                F, G, Ht);
  attn_scores_kernel<<<8, 128, 0, stream>>>(F, G, BetaT);
  attn_apply_kernel<<<dim3(256, 8), 128, 0, stream>>>(BetaT, Ht, O);
  out_kernel<<<2048, 256, 0, stream>>>(O, WlT, blv, x, gamma, betaL,
                                       (float*)d_out);
}
